// QuantizedLinear_30777735643295
// MI455X (gfx1250) — matrix operand order repaired and checked by reading, not compiled, not run on hardware
//
#include <hip/hip_runtime.h>

// Types for WMMA operands / vector memory ops
typedef __attribute__((ext_vector_type(16))) _Float16 v16h;
typedef __attribute__((ext_vector_type(8)))  _Float16 v8h;
typedef __attribute__((ext_vector_type(4)))  _Float16 v4h;
typedef __attribute__((ext_vector_type(8)))  float    v8f;
typedef __attribute__((ext_vector_type(4)))  float    v4f;
typedef __attribute__((ext_vector_type(4)))  int      v4i;

union Frag16 { v16h v; v8h h[2]; };

#define BM 256
#define BN 128
#define BK 64
#define LDT 72   // LDS row stride in halves (BK + 8 pad -> 144B rows)

// ---------------------------------------------------------------------------
// CDNA5 async global->LDS copy (ASYNCcnt-tracked, no VGPR round trip).
// GVS form: LDS-addr VGPR, 32-bit byte-offset VGPR, 64-bit SGPR base.
// ---------------------------------------------------------------------------
__device__ __forceinline__ void async_copy_b128(unsigned lds_off,
                                                const void* base, int goff) {
    asm volatile("global_load_async_to_lds_b128 %0, %1, %2"
                 :: "v"(lds_off), "v"(goff), "s"(base)
                 : "memory");
}
__device__ __forceinline__ unsigned lds_addr(const void* p) {
    return (unsigned)(unsigned long long)p;   // flat LDS addr[31:0] == LDS offset
}

// ---------------------------------------------------------------------------
// Kernel 1: convert int32 weight codes (0..15) -> f16 (exact). W is [N, K].
// ---------------------------------------------------------------------------
__global__ __launch_bounds__(256) void qlin_convert_w(const int* __restrict__ q,
                                                      _Float16* __restrict__ wh,
                                                      long long n) {
    long long i = ((long long)blockIdx.x * 256 + threadIdx.x) * 4;
    if (i + 4 <= n) {
        v4i v = __builtin_nontemporal_load((const v4i*)(q + i));
        v4h hv = { (_Float16)v.x, (_Float16)v.y, (_Float16)v.z, (_Float16)v.w };
        *(v4h*)(wh + i) = hv;   // regular store: keep f16 weights L2-resident
    }
}

// ---------------------------------------------------------------------------
// Kernel 2: convert x fp32 -> f16 and compute rowsum over the rounded f16
// values. One 256-thread block per row of x[M, K].
// ---------------------------------------------------------------------------
__global__ __launch_bounds__(256) void qlin_convert_x(const float* __restrict__ x,
                                                      _Float16* __restrict__ xh,
                                                      float* __restrict__ rowsum,
                                                      int K) {
    __shared__ float red[256];
    const int row = blockIdx.x;
    const float* xr = x + (size_t)row * K;
    _Float16* xo = xh + (size_t)row * K;
    float s = 0.f;
    for (int k = threadIdx.x * 4; k < K; k += 256 * 4) {
        v4f v = __builtin_nontemporal_load((const v4f*)(xr + k));
        _Float16 h0 = (_Float16)v.x, h1 = (_Float16)v.y;
        _Float16 h2 = (_Float16)v.z, h3 = (_Float16)v.w;
        s += (float)h0 + (float)h1 + (float)h2 + (float)h3;
        v4h hv = { h0, h1, h2, h3 };
        *(v4h*)(xo + k) = hv;   // regular store: keep f16 x L2-resident
    }
    red[threadIdx.x] = s;
    __syncthreads();
    for (int off = 128; off > 0; off >>= 1) {
        if ((int)threadIdx.x < off) red[threadIdx.x] += red[threadIdx.x + off];
        __syncthreads();
    }
    if (threadIdx.x == 0) rowsum[row] = red[0];
}

// ---------------------------------------------------------------------------
// Kernel 3: C[M,N] = Ah[M,K] * Bh[N,K]^T via v_wmma_f32_16x16x32_f16.
// 256 threads = 8 wave32 arranged 4(M) x 2(N); per-wave tile 64x64 = 4x4
// fragments (16 WMMA per 32-deep k-step, each frag reused 4x).
// Staging via async global->LDS copies, double-buffered.
// Epilogue: out = scale[n] * (acc - zp[n]*rowsum[m]), nontemporal stores.
// ---------------------------------------------------------------------------
__global__ __launch_bounds__(256) void qlin_wmma_gemm(
    const _Float16* __restrict__ Ah,   // [M,K] f16
    const _Float16* __restrict__ Bh,   // [N,K] f16
    const float* __restrict__ rowsum,  // [M]
    const float* __restrict__ scale,   // [N]
    const float* __restrict__ zp,      // [N]
    float* __restrict__ out,           // [M,N]
    int M, int N, int K)
{
    __shared__ __align__(16) _Float16 As[2][BM * LDT];
    __shared__ __align__(16) _Float16 Bs[2][BN * LDT];

    const int tid  = threadIdx.x;
    const int lane = tid & 31;
    const int w    = tid >> 5;     // wave id 0..7
    const int wm   = w >> 1;       // 0..3 -> 64-row slab
    const int wn   = w & 1;        // 0..1 -> 64-col slab
    const int r    = lane & 15;
    const int h    = lane >> 4;

    const int nblk = N / BN;
    const int m0 = (blockIdx.x / nblk) * BM;
    const int n0 = (blockIdx.x % nblk) * BN;

    // Issue one stage of async copies: A 256x64 (8 chunks/thread),
    // B 128x64 (4 chunks/thread); 12 ASYNCcnt events per wave per stage.
    auto issue_stage = [&](int buf, int k0) {
        #pragma unroll
        for (int p = 0; p < 8; ++p) {
            const int e = (p * 256 + tid) * 8;
            const int row = e >> 6, kk = e & (BK - 1);
            async_copy_b128(lds_addr(&As[buf][row * LDT + kk]), Ah,
                            (int)(((size_t)(m0 + row) * K + k0 + kk) * 2));
        }
        #pragma unroll
        for (int p = 0; p < 4; ++p) {
            const int e = (p * 256 + tid) * 8;
            const int row = e >> 6, kk = e & (BK - 1);
            async_copy_b128(lds_addr(&Bs[buf][row * LDT + kk]), Bh,
                            (int)(((size_t)(n0 + row) * K + k0 + kk) * 2));
        }
    };

    v8f acc[4][4] = {};

    issue_stage(0, 0);

    for (int k0 = 0; k0 < K; k0 += BK) {
        const int buf = (k0 / BK) & 1;
        if (k0 + BK < K) {
            issue_stage(buf ^ 1, k0 + BK);
            // 12 newer copies outstanding; async loads complete in order,
            // so <=12 drains exactly the current stage.
            asm volatile("s_wait_asynccnt 12" ::: "memory");
        } else {
            asm volatile("s_wait_asynccnt 0" ::: "memory");
        }
        __syncthreads();   // make all waves' async LDS writes visible

        #pragma unroll
        for (int ks = 0; ks < BK; ks += 32) {
            Frag16 af[4], bf[4];
            // A fragment (16x32 f16): lanes 0-15: K 0..7 & 16..23;
            // lanes 16-31: K 8..15 & 24..31 (ISA 7.12.2).
            #pragma unroll
            for (int i = 0; i < 4; ++i) {
                const _Float16* base = &As[buf][(wm * 64 + i * 16 + r) * LDT + ks];
                af[i].h[0] = *(const v8h*)(base + h * 8);
                af[i].h[1] = *(const v8h*)(base + 16 + h * 8);
            }
            // B fragment (32x16 f16): lanes 0-15: K 0..7 & 16..23; lanes 16-31:
            // K 8..15 & 24..31; column = lane&15 (same k order as A).
            #pragma unroll
            for (int j = 0; j < 4; ++j) {
                const _Float16* base = &Bs[buf][(wn * 64 + j * 16 + r) * LDT + ks];
                bf[j].h[0] = *(const v8h*)(base + h * 8);
                bf[j].h[1] = *(const v8h*)(base + 16 + h * 8);
            }
            #pragma unroll
            for (int i = 0; i < 4; ++i)
                #pragma unroll
                for (int j = 0; j < 4; ++j)
                    acc[i][j] = __builtin_amdgcn_wmma_f32_16x16x32_f16(
                        false, af[i].v, false, bf[j].v,
                        (short)0, acc[i][j], false, false);
        }
        __syncthreads();   // all reads of `buf` done before it is rewritten
    }

    // --- epilogue: out = scale[n] * (acc - zp[n]*rowsum[m]) ---
    // C/D layout: n = lane&15, m = (lane>>4)*8 + vgpr (ISA 7.12.2)
    #pragma unroll
    for (int j = 0; j < 4; ++j) {
        const int gn = n0 + wn * 64 + j * 16 + r;
        const float sc  = scale[gn];
        const float zpv = zp[gn];
        #pragma unroll
        for (int i = 0; i < 4; ++i) {
            const int gmb = m0 + wm * 64 + i * 16 + h * 8;
            #pragma unroll
            for (int v = 0; v < 8; ++v) {
                const float res = sc * (acc[i][j][v] - zpv * rowsum[gmb + v]);
                __builtin_nontemporal_store(res, &out[(size_t)(gmb + v) * N + gn]);
            }
        }
    }
}

// ---------------------------------------------------------------------------
// Host launcher. Workspace: [xh f16 M*K][wh f16 N*K][rowsum f32 M] (~96 MB).
// ---------------------------------------------------------------------------
extern "C" void kernel_launch(void* const* d_in, const int* in_sizes, int n_in,
                              void* d_out, int out_size, void* d_ws, size_t ws_size,
                              hipStream_t stream) {
    const float* x     = (const float*)d_in[0];
    const float* scale = (const float*)d_in[1];
    const float* zp    = (const float*)d_in[2];
    const int*   qw    = (const int*)d_in[3];   // integer inputs arrive as int32
    float* out = (float*)d_out;

    const int N = in_sizes[1];
    const long long totW = (long long)in_sizes[3];
    const int K = (int)(totW / N);
    const int M = (int)((long long)in_sizes[0] / K);

    char* ws = (char*)d_ws;
    _Float16* xh = (_Float16*)ws;
    _Float16* wh = (_Float16*)(ws + (size_t)M * K * sizeof(_Float16));
    float* rowsum = (float*)(ws + (size_t)M * K * sizeof(_Float16)
                                + (size_t)N * K * sizeof(_Float16));

    {
        long long nthreads4 = totW / 4;
        int grid = (int)((nthreads4 + 255) / 256);
        qlin_convert_w<<<grid, 256, 0, stream>>>(qw, wh, totW);
    }
    qlin_convert_x<<<M, 256, 0, stream>>>(x, xh, rowsum, K);

    const int grid = (M / BM) * (N / BN);
    qlin_wmma_gemm<<<grid, 256, 0, stream>>>(xh, wh, rowsum, scale, zp, out, M, N, K);
}
